// CausalAttentionHead_31456340476427
// MI455X (gfx1250) — matrix operand order repaired and checked by reading, not compiled, not run on hardware
//
#include <hip/hip_runtime.h>
#include <hip/hip_bf16.h>

// Causal single-head attention for MI455X (gfx1250, wave32, WMMA).
//   B=8, T=2048, C=1024, H=64.  fp32 in/out, bf16 WMMA internally (f32 accum).
//
// Pipeline (all on `stream`):
//   K0 wpack    : W{q,k,v} f32 -> bf16, pre-swizzled into WMMA B-operand order
//   K1 qkv_proj : q,k,v = x @ W (q scaled by C^-0.5), bf16 planes in ws
//   K2 vpack    : v plane -> V^T pre-swizzled into WMMA A-operand order
//   K3 attn     : flash attention, transposed formulation (S^T = K Q^T, O^T = V^T P^T)
//
// ws layout: [ q|k|v bf16 : 6 MB ][ wpk : 384 KB ][ vtp : 2 MB ]

#define B_ 8
#define T_ 2048
#define C_ 1024
#define H_ 64

#define QKV_BYTES (3u * B_ * T_ * H_ * 2u)        // 6 MB
#define WPK_BYTES (3u * 32u * 4u * 32u * 8u * 4u) // 384 KB
#define VTP_DW_PER_B ((size_t)(T_ / 32) * 4 * 32 * 8)

typedef __attribute__((ext_vector_type(16))) __bf16 bf16x16;
typedef __attribute__((ext_vector_type(2)))  __bf16 bf16x2;
typedef __attribute__((ext_vector_type(8)))  float  floatx8;

union ABf {               // one WMMA A/B operand: 16 bf16 halves = 8 dwords / lane
    bf16x16      v;
    unsigned int u[8];
};

// f32 -> bf16 through the compiler's native conversion (hw cvt on gfx1250)
__device__ __forceinline__ unsigned short f2bf(float f) {
    union { __bf16 h; unsigned short s; } t;
    t.h = (__bf16)f;
    return t.s;
}
__device__ __forceinline__ unsigned int pk(float lo, float hi) {
    union { bf16x2 h; unsigned int u; } t;
    t.h = bf16x2{(__bf16)lo, (__bf16)hi};
    return t.u;
}

__device__ __forceinline__ floatx8 wmma_bf16(const ABf& a, const ABf& b, floatx8 c) {
    return __builtin_amdgcn_wmma_f32_16x16x32_bf16(
        /*neg_a=*/false, a.v, /*neg_b=*/false, b.v,
        /*c_mod=*/(short)0, c, /*reuse_a=*/false, /*reuse_b=*/false);
}

__device__ __forceinline__ void load_op(ABf& d, const unsigned int* p) {
    *(uint4*)&d.u[0] = *(const uint4*)(p);
    *(uint4*)&d.u[4] = *(const uint4*)(p + 4);
}

// ---------------------------------------------------------------------------
// K0: W f32 -> bf16 packed as B-operands. Job = (m, ktile, ntile), 384 waves.
// Slot: wpk[(((m*32 + kt)*4 + nt)*32 + lane)*8 + i]
// ---------------------------------------------------------------------------
__global__ __launch_bounds__(256) void wpack_kernel(
    const float* __restrict__ Wq, const float* __restrict__ Wk,
    const float* __restrict__ Wv, unsigned int* __restrict__ wpk)
{
    const int job  = (blockIdx.x * 256 + threadIdx.x) >> 5;   // 0..383
    const int lane = threadIdx.x & 31;
    const int nt = job & 3, kt = (job >> 2) & 31, m = job >> 7;
    if (m >= 3) return;
    const float* W = (m == 0) ? Wq : ((m == 1) ? Wk : Wv);

    const int col = lane & 15;
    const int kb2 = (lane < 16) ? 0 : 8;                      // B-layout K base
    const float* wp = W + (size_t)(kt * 32 + kb2) * H_ + nt * 16 + col;

    unsigned int u[8];
    #pragma unroll
    for (int i = 0; i < 8; ++i)
        u[i] = pk(wp[(size_t)((i < 4) ? 2 * i : 2 * i + 8) * H_], wp[(size_t)((i < 4) ? 2 * i + 1 : 2 * i + 9) * H_]);

    unsigned int* dst = wpk + (((size_t)(m * 32 + kt) * 4 + nt) * 32 + lane) * 8;
    *(uint4*)dst       = uint4{u[0], u[1], u[2], u[3]};
    *(uint4*)(dst + 4) = uint4{u[4], u[5], u[6], u[7]};
}

// ---------------------------------------------------------------------------
// K1: fused QKV projection. One wave per 16-row tile of x; A built once per
// K-step. All 12 prepacked B operands are loaded into distinct registers
// before the 12 WMMAs so loads overlap matrix issue (no loadcnt-0 stalls).
// ---------------------------------------------------------------------------
__global__ __launch_bounds__(256) void qkv_proj_kernel(
    const float* __restrict__ x,
    const unsigned int* __restrict__ wpk,
    unsigned short* __restrict__ qkv)
{
    const int wave = (blockIdx.x * 256 + threadIdx.x) >> 5;
    const int lane = threadIdx.x & 31;
    if (wave >= (B_ * T_) / 16) return;      // never taken

    const float* xt = x + (size_t)wave * 16 * C_;
    const int row = lane & 15;               // A-layout: M = lane%16
    const int h0  = (lane < 16) ? 0 : 8;     // A-layout K-group base
    const int col = lane & 15;

    floatx8 c[12] = {};                      // [m*4 + nt]

    for (int kb = 0; kb < C_; kb += 32) {
        const int kt = kb >> 5;
        const unsigned int* wb = wpk + (size_t)kt * 1024 + (size_t)lane * 8;

        // ---- issue all 12 B-operand loads first (24x b128, distinct regs)
        ABf bop[12];
        #pragma unroll
        for (int j = 0; j < 12; ++j) {
            const int m = j >> 2, nt = j & 3;
            load_op(bop[j], wb + (size_t)m * 32768 + (size_t)nt * 256);
        }

        // ---- A operand: x[row, kb + {h0..h0+7, h0+16..h0+23}] f32 -> bf16
        ABf a;
        const float* xr = xt + (size_t)row * C_ + kb;
        float4 f0 = *(const float4*)(xr + h0);
        float4 f1 = *(const float4*)(xr + h0 + 4);
        float4 f2 = *(const float4*)(xr + h0 + 16);
        float4 f3 = *(const float4*)(xr + h0 + 20);
        a.u[0] = pk(f0.x, f0.y); a.u[1] = pk(f0.z, f0.w);
        a.u[2] = pk(f1.x, f1.y); a.u[3] = pk(f1.z, f1.w);
        a.u[4] = pk(f2.x, f2.y); a.u[5] = pk(f2.z, f2.w);
        a.u[6] = pk(f3.x, f3.y); a.u[7] = pk(f3.z, f3.w);

        #pragma unroll
        for (int j = 0; j < 12; ++j)
            c[j] = wmma_bf16(a, bop[j], c[j]);
    }

    // ---- store C layout (row = r + 8*(lane/16), col = lane%16) as bf16
    const int rbase = (lane >> 4) * 8;
    #pragma unroll
    for (int m = 0; m < 3; ++m) {
        const float scale = (m == 0) ? 0.03125f : 1.0f;   // C^-0.5 folded into q
        unsigned short* outp = qkv + (size_t)m * (B_ * T_ * H_)
                                   + (size_t)wave * 16 * H_;
        #pragma unroll
        for (int nt = 0; nt < 4; ++nt)
            #pragma unroll
            for (int r = 0; r < 8; ++r)
                outp[(size_t)(rbase + r) * H_ + nt * 16 + col] =
                    f2bf(c[m * 4 + nt][r] * scale);
    }
}

// ---------------------------------------------------------------------------
// K2: v plane -> V^T A-operand order. Job = (batch, 32-key block), 512 waves.
// Slot: vtp[(((b*(T/32) + blk)*4 + nt)*32 + lane)*8 + i]
// ---------------------------------------------------------------------------
__global__ __launch_bounds__(256) void vpack_kernel(
    const unsigned short* __restrict__ vplane,
    unsigned int* __restrict__ vtp)
{
    const int job  = (blockIdx.x * 256 + threadIdx.x) >> 5;   // 0..511
    const int lane = threadIdx.x & 31;
    const int b   = job >> 6;
    const int blk = job & 63;
    if (b >= B_) return;

    const int col = lane & 15;
    const int h0  = (lane >> 4) * 16;                         // A-layout K-group base
    const unsigned short* vb0 = vplane + (size_t)b * T_ * H_ + (size_t)blk * 32 * H_;

    #pragma unroll
    for (int nt = 0; nt < 4; ++nt) {
        const unsigned short* vb = vb0 + nt * 16 + col;
        unsigned int u[8];
        #pragma unroll
        for (int i = 0; i < 8; ++i) {
            const int kk = h0 + 2 * i;
            const unsigned int lo = vb[(size_t)kk * H_];
            const unsigned int hi = vb[(size_t)(kk + 1) * H_];
            u[i] = lo | (hi << 16);
        }
        unsigned int* dst = vtp + (((size_t)(b * 64 + blk) * 4 + nt) * 32 + lane) * 8;
        *(uint4*)dst       = uint4{u[0], u[1], u[2], u[3]};
        *(uint4*)(dst + 4) = uint4{u[4], u[5], u[6], u[7]};
    }
}

// ---------------------------------------------------------------------------
// K3: causal flash attention, one wave per (batch, 16-query tile).
//   S^T = K_tile(16x64) * Q^T(64x16)      [2 WMMAs per 16-key sub-tile]
//   O^T += V^T(16x32) * P^T(32x16)        [4 WMMAs per 32-key block]
// All 16 b128 loads of a key block are issued up front; the V^T operands
// retire behind the S^T WMMAs and the softmax VALU work.
// ---------------------------------------------------------------------------
__global__ __launch_bounds__(256) void attn_kernel(
    const unsigned short* __restrict__ qkv,
    const unsigned int* __restrict__ vtp,
    float* __restrict__ out)
{
    const int wave = (blockIdx.x * 256 + threadIdx.x) >> 5;
    const int lane = threadIdx.x & 31;
    const int tilesPerBatch = T_ / 16;
    const int b  = wave / tilesPerBatch;
    const int qt = wave % tilesPerBatch;
    if (b >= B_) return;                     // never taken

    const unsigned short* qp = qkv + (size_t)b * T_ * H_;
    const unsigned short* kp = qkv + (size_t)(B_ * T_ * H_) + (size_t)b * T_ * H_;
    const unsigned int*   vt = vtp + (size_t)b * VTP_DW_PER_B + (size_t)lane * 8;

    const int q0   = qt * 16;
    const int lcol = lane & 15;              // B/C-layout column (the query)
    const int half = lane >> 4;
    const int kb2  = half * 16;              // B-layout K base
    const int h0   = half * 8;               // A-layout K-group base
    const int myq  = q0 + lcol;

    // ---- Q^T B-operands (64-K split into two 32-K chunks); contiguous 32B loads
    ABf bq0, bq1;
    {
        const unsigned short* qr = qp + (size_t)(q0 + lcol) * H_;
        *(uint4*)&bq0.u[0] = *(const uint4*)(qr + kb2);
        *(uint4*)&bq0.u[4] = *(const uint4*)(qr + kb2 + 8);
        *(uint4*)&bq1.u[0] = *(const uint4*)(qr + 32 + kb2);
        *(uint4*)&bq1.u[4] = *(const uint4*)(qr + 32 + kb2 + 8);
    }

    floatx8 ot[4] = {};                      // O^T: 4 h-tiles of 16x16 f32
    float m_i = -1e30f, l_i = 0.0f;

    for (int kb0 = 0; kb0 < q0 + 16; kb0 += 32) {
        // ---- issue all loads for this key block up front
        ABf av[4];                           // V^T operands (prepacked)
        const int blk = kb0 >> 5;
        #pragma unroll
        for (int nt = 0; nt < 4; ++nt)
            load_op(av[nt], vt + ((size_t)blk * 4 + nt) * 256);

        ABf ak[2][2];                        // K-tile A operands
        #pragma unroll
        for (int t = 0; t < 2; ++t) {
            const unsigned short* kr = kp + (size_t)(kb0 + t * 16 + lcol) * H_;
            *(uint4*)&ak[t][0].u[0] = *(const uint4*)(kr + kb2);
            *(uint4*)&ak[t][0].u[4] = *(const uint4*)(kr + kb2 + 8);
            *(uint4*)&ak[t][1].u[0] = *(const uint4*)(kr + kb2 + 32);
            *(uint4*)&ak[t][1].u[4] = *(const uint4*)(kr + kb2 + 40);
        }

        // ---- S^T = K Q^T (+ causal mask)
        float p[2][8];
        float mx = -1e30f;
        #pragma unroll
        for (int t = 0; t < 2; ++t) {
            floatx8 st = {};
            st = wmma_bf16(ak[t][0], bq0, st);
            st = wmma_bf16(ak[t][1], bq1, st);
            #pragma unroll
            for (int r = 0; r < 8; ++r) {
                const int key = kb0 + t * 16 + r + 8 * half;
                const float s = (key <= myq) ? st[r] : -1e30f;
                p[t][r] = s;
                mx = fmaxf(mx, s);
            }
        }

        // ---- online softmax; partner lane (xor 16) holds same query, other keys
        mx = fmaxf(mx, __shfl_xor(mx, 16, 32));
        const float m_new = fmaxf(m_i, mx);
        const float alpha = __expf(m_i - m_new);
        float rsum = 0.0f;
        #pragma unroll
        for (int t = 0; t < 2; ++t)
            #pragma unroll
            for (int r = 0; r < 8; ++r) {
                const float e = __expf(p[t][r] - m_new);
                p[t][r] = e;
                rsum += e;
            }
        rsum += __shfl_xor(rsum, 16, 32);
        l_i = l_i * alpha + rsum;
        m_i = m_new;
        #pragma unroll
        for (int nt = 0; nt < 4; ++nt)
            #pragma unroll
            for (int r = 0; r < 8; ++r)
                ot[nt][r] *= alpha;

        // ---- glue the two S^T C-layout tiles into the P^T B-operand (32K x 16N)
        float jl[8], jh[8];
        #pragma unroll
        for (int r = 0; r < 8; ++r) {
            const float send = half ? p[0][r] : p[1][r];
            const float recv = __shfl_xor(send, 16, 32);
            jl[r] = half ? recv    : p[0][r];
            jh[r] = half ? p[1][r] : recv;
        }
        ABf bp;
        bp.u[0] = pk(jl[0], jl[1]); bp.u[1] = pk(jl[2], jl[3]);
        bp.u[2] = pk(jl[4], jl[5]); bp.u[3] = pk(jl[6], jl[7]);
        bp.u[4] = pk(jh[0], jh[1]); bp.u[5] = pk(jh[2], jh[3]);
        bp.u[6] = pk(jh[4], jh[5]); bp.u[7] = pk(jh[6], jh[7]);

        // ---- O^T += V^T * P^T (V^T operands already resident)
        #pragma unroll
        for (int nt = 0; nt < 4; ++nt)
            ot[nt] = wmma_bf16(av[nt], bp, ot[nt]);
    }

    // ---- normalize and store: out[b, q, h] with h = nt*16 + r + 8*half
    const float inv = 1.0f / l_i;
    float* ob = out + ((size_t)b * T_ + myq) * H_;
    #pragma unroll
    for (int nt = 0; nt < 4; ++nt)
        #pragma unroll
        for (int r = 0; r < 8; ++r)
            ob[nt * 16 + r + 8 * half] = ot[nt][r] * inv;
}

// ---------------------------------------------------------------------------
extern "C" void kernel_launch(void* const* d_in, const int* in_sizes, int n_in,
                              void* d_out, int out_size, void* d_ws, size_t ws_size,
                              hipStream_t stream)
{
    const float* x  = (const float*)d_in[0];
    const float* Wq = (const float*)d_in[1];
    const float* Wk = (const float*)d_in[2];
    const float* Wv = (const float*)d_in[3];
    float* out = (float*)d_out;

    unsigned short* qkv = (unsigned short*)d_ws;
    unsigned int*   wpk = (unsigned int*)((char*)d_ws + QKV_BYTES);
    unsigned int*   vtp = (unsigned int*)((char*)d_ws + QKV_BYTES + WPK_BYTES);

    wpack_kernel<<<384 / 8, 256, 0, stream>>>(Wq, Wk, Wv, wpk);

    const int projWaves = (B_ * T_) / 16;                 // 1024
    qkv_proj_kernel<<<projWaves / 8, 256, 0, stream>>>(x, wpk, qkv);

    vpack_kernel<<<512 / 8, 256, 0, stream>>>(qkv + (size_t)2 * B_ * T_ * H_, vtp);

    const int attnWaves = B_ * (T_ / 16);                 // 1024
    attn_kernel<<<attnWaves / 8, 256, 0, stream>>>(qkv, vtp, out);
}
